// VectorQuantizer_53008486367576
// MI455X (gfx1250) — matrix operand order repaired and checked by reading, not compiled, not run on hardware
//
#include <hip/hip_runtime.h>
#include <hip/hip_bf16.h>

typedef __attribute__((ext_vector_type(16))) __bf16 v16bf;
typedef __attribute__((ext_vector_type(8)))  float  v8f;

#define NUM_EMB 512
#define EMB_DIM 64
#define DHW     65536        // 16*64*64 spatial per batch
#define NROWS   262144       // 4 * DHW flat vectors
#define NPART   16384        // NROWS / 16 row-tiles
#define TPB     4            // row-tiles (waves) per block

// workspace byte offsets
#define WS_BHI   0           // 32768 bf16 = 64 KB packed B hi fragments
#define WS_BLO   65536       // 64 KB packed B lo fragments
#define WS_E2    131072      // 512 floats ||e||^2
#define WS_USED  133120      // 512 uints used flags
#define WS_PART  135168      // 16384 floats per-tile loss partials

// ---------------- prep: ||e||^2 + zero used flags ----------------
__global__ void vq_prep_small(const float* __restrict__ emb, void* ws) {
    float*    e2   = (float*)((char*)ws + WS_E2);
    unsigned* used = (unsigned*)((char*)ws + WS_USED);
    int t = blockIdx.x * 256 + threadIdx.x;
    if (t < NUM_EMB) {
        float s = 0.f;
        #pragma unroll 8
        for (int k = 0; k < EMB_DIM; ++k) { float v = emb[t * EMB_DIM + k]; s += v * v; }
        e2[t] = s;
        used[t] = 0u;
    }
}

// ---------------- prep: pack emb into WMMA B-fragment layout (hi/lo bf16) ----
// element index: idx = ((ct*2 + kc)*32 + lane)*16 + e
// code n = ct*16 + (lane&15);  dim k = kc*32 + e + (lane>=16 ? 16 : 0)
__global__ void vq_prep_pack(const float* __restrict__ emb, void* ws) {
    __bf16* Bhi = (__bf16*)((char*)ws + WS_BHI);
    __bf16* Blo = (__bf16*)((char*)ws + WS_BLO);
    int idx  = blockIdx.x * 256 + threadIdx.x;   // 0 .. 32767
    int e    = idx & 15;
    int lane = (idx >> 4) & 31;
    int kc   = (idx >> 9) & 1;
    int ct   = idx >> 10;
    int n    = (ct << 4) + (lane & 15);
    int k    = (kc << 5) + e + ((lane >= 16) ? 16 : 0);
    float v  = emb[n * EMB_DIM + k];
    __bf16 hi = (__bf16)v;
    Bhi[idx] = hi;
    Blo[idx] = (__bf16)(v - (float)hi);
}

// ---------------- main: distances via split-bf16 WMMA, argmin, gather -------
__global__ __launch_bounds__(128)
void vq_main(const float* __restrict__ z, const float* __restrict__ emb,
             void* ws, float* __restrict__ out) {
    const __bf16* Bhig = (const __bf16*)((char*)ws + WS_BHI);
    const __bf16* Blog = (const __bf16*)((char*)ws + WS_BLO);
    const float*  e2   = (const float*)((char*)ws + WS_E2);
    unsigned*     used = (unsigned*)((char*)ws + WS_USED);
    float*        part = (float*)((char*)ws + WS_PART);

    __shared__ float As[TPB][EMB_DIM * 16];   // [c][m] fp32 staged tile
    __shared__ float rmin[TPB][16][16];
    __shared__ int   ridx[TPB][16][16];
    __shared__ int   rowIdx[TPB][16];

    const int w    = threadIdx.x >> 5;
    const int lane = threadIdx.x & 31;
    const int tile = blockIdx.x * TPB + w;
    const int gn0  = tile << 4;               // first flat row of tile
    const int b    = gn0 >> 16;               // DHW = 65536
    const int sp0  = gn0 & (DHW - 1);
    const float* zbase = z + ((size_t)b * EMB_DIM) * DHW + sp0;

    // phase 1: cooperative coalesced load of 64x16 tile into LDS
    #pragma unroll
    for (int it = 0; it < 8; ++it) {
        int q = it * 32 + lane;
        int c = q >> 2, f = (q & 3) << 2;
        float4 v = *reinterpret_cast<const float4*>(zbase + (size_t)c * DHW + f);
        *reinterpret_cast<float4*>(&As[w][c * 16 + f]) = v;
    }
    __syncthreads();

    // phase 2: build A fragments (16-bit A 16x32 lane layout), hi/lo split
    const int M    = lane & 15;
    const int koff = (lane >> 4) << 4;        // +16 K-offset for upper half-wave
    v16bf Ahi[2], Alo[2];
    #pragma unroll
    for (int kc = 0; kc < 2; ++kc) {
        #pragma unroll
        for (int e = 0; e < 16; ++e) {
            int K = e + koff;
            float x = As[w][((kc << 5) + K) * 16 + M];
            __bf16 hi = (__bf16)x;
            Ahi[kc][e] = hi;
            Alo[kc][e] = (__bf16)(x - (float)hi);
        }
    }

    float minv[8]; int mini[8];
    #pragma unroll
    for (int v = 0; v < 8; ++v) { minv[v] = 3.4e38f; mini[v] = 0; }

    const v16bf* Bh = reinterpret_cast<const v16bf*>(Bhig);
    const v16bf* Bl = reinterpret_cast<const v16bf*>(Blog);

    // phase 3: 32 column tiles of 16 codes; 6 WMMAs each (error-compensated)
    for (int ct = 0; ct < 32; ++ct) {
        v8f acc = {};
        #pragma unroll
        for (int kc = 0; kc < 2; ++kc) {
            v16bf bh = Bh[((ct << 1) + kc) * 32 + lane];
            v16bf bl = Bl[((ct << 1) + kc) * 32 + lane];
            acc = __builtin_amdgcn_wmma_f32_16x16x32_bf16(false, Ahi[kc], false, bh, (short)0, acc, false, false);
            acc = __builtin_amdgcn_wmma_f32_16x16x32_bf16(false, Alo[kc], false, bh, (short)0, acc, false, false);
            acc = __builtin_amdgcn_wmma_f32_16x16x32_bf16(false, Ahi[kc], false, bl, (short)0, acc, false, false);
        }
        int n = (ct << 4) + (lane & 15);
        float en2 = e2[n];
        #pragma unroll
        for (int v = 0; v < 8; ++v) {
            float d = en2 - 2.0f * acc[v];   // ||x||^2 dropped (row-constant)
            if (d < minv[v]) { minv[v] = d; mini[v] = n; }
        }
    }

    // phase 4: cross-lane argmin. C/D layout: row Mv = v + 8*(lane>=16)
    #pragma unroll
    for (int v = 0; v < 8; ++v) {
        int Mv = v + ((lane >> 4) << 3);
        rmin[w][Mv][lane & 15] = minv[v];
        ridx[w][Mv][lane & 15] = mini[v];
    }
    __syncthreads();
    if (lane < 16) {
        int m = lane;
        float best = rmin[w][m][0]; int bi = ridx[w][m][0];
        #pragma unroll
        for (int j = 1; j < 16; ++j) {
            float d = rmin[w][m][j];
            if (d < best) { best = d; bi = ridx[w][m][j]; }
        }
        rowIdx[w][m] = bi;
        used[bi] = 1u;
    }
    __syncthreads();

    // phase 5: gather z_q from fp32 codebook, write out, accumulate loss
    float lsum = 0.f;
    #pragma unroll 4
    for (int it = 0; it < 32; ++it) {
        int q = it * 32 + lane;
        int m = q & 15, c = q >> 4;
        float val  = emb[rowIdx[w][m] * EMB_DIM + c];
        float diff = val - As[w][c * 16 + m];
        lsum += diff * diff;
        out[((size_t)b * EMB_DIM + c) * DHW + sp0 + m] = val;
    }
    #pragma unroll
    for (int off = 16; off > 0; off >>= 1)
        lsum += __shfl_xor(lsum, off, 32);
    if (lane == 0) part[tile] = lsum;
}

// ---------------- finalize: deterministic loss + unique-code count ----------
__global__ __launch_bounds__(256)
void vq_finalize(void* ws, float* __restrict__ out) {
    const float*    part = (const float*)((char*)ws + WS_PART);
    const unsigned* used = (const unsigned*)((char*)ws + WS_USED);
    __shared__ float ssum[256];
    __shared__ int   scnt[256];
    float s = 0.f;
    for (int i = threadIdx.x; i < NPART; i += 256) s += part[i];
    int c = 0;
    for (int i = threadIdx.x; i < NUM_EMB; i += 256) c += (used[i] != 0u) ? 1 : 0;
    ssum[threadIdx.x] = s; scnt[threadIdx.x] = c;
    __syncthreads();
    for (int off = 128; off > 0; off >>= 1) {
        if (threadIdx.x < off) {
            ssum[threadIdx.x] += ssum[threadIdx.x + off];
            scnt[threadIdx.x] += scnt[threadIdx.x + off];
        }
        __syncthreads();
    }
    if (threadIdx.x == 0) {
        // codebook_loss + 0.25*commitment_loss == 1.25 * mean((z_q - z)^2)
        out[16777216] = 1.25f * ssum[0] / 16777216.0f;
        out[16777217] = (float)scnt[0];
    }
}

extern "C" void kernel_launch(void* const* d_in, const int* in_sizes, int n_in,
                              void* d_out, int out_size, void* d_ws, size_t ws_size,
                              hipStream_t stream) {
    const float* z   = (const float*)d_in[0];
    const float* emb = (const float*)d_in[1];
    float* out = (float*)d_out;

    vq_prep_small<<<2, 256, 0, stream>>>(emb, d_ws);
    vq_prep_pack <<<128, 256, 0, stream>>>(emb, d_ws);
    vq_main      <<<NPART / TPB, 32 * TPB, 0, stream>>>(z, emb, d_ws, out);
    vq_finalize  <<<1, 256, 0, stream>>>(d_ws, out);
}
